// GAT_28106265985526
// MI455X (gfx1250) — matrix operand order repaired and checked by reading, not compiled, not run on hardware
//
#include <hip/hip_runtime.h>
#include <hip/hip_bf16.h>

typedef __attribute__((ext_vector_type(16))) _Float16 v16h;
typedef __attribute__((ext_vector_type(8)))  float    v8f;

#define B_  8
#define N_  1024
#define F_  256
#define D_  64
#define H_  8
#define C_  256
#define HD_ 512
#define ALPHA_ 0.2f

// ---- WMMA helper (codegen-confirmed builtin) -------------------------------
__device__ __forceinline__ v8f wmma_f16(v16h a, v16h b, v8f c) {
  return __builtin_amdgcn_wmma_f32_16x16x32_f16(false, a, false, b, (short)0, c,
                                                false, false);
}

// A-operand (16x32, 16-bit), k order used in this file: lane l -> M = l&15, K = (l>>4)*16 + {0..15}
// B-operand (32x16, 16-bit), same k order as A: lane l -> N = l&15, K = (l>>4)*16 + {0..15}
//   (one contiguous 32B v16h load from a [*, K-major] transposed array)
// C/D (16x16 f32, 8 VGPRs): elem r, lane l -> M = r + (l>>4)*8, N = l&15

// ============================================================================
// K0: prep — transpose+convert weights to f16 K-major layouts
// ============================================================================
__global__ __launch_bounds__(256) void k0_prep(const float* __restrict__ W,
                                               const float* __restrict__ Wo,
                                               _Float16* __restrict__ WT,
                                               _Float16* __restrict__ WoT) {
  int t = blockIdx.x * blockDim.x + threadIdx.x;
  int nth = gridDim.x * blockDim.x;
  // WT[h][d][f] = W[h][f][d]
  for (int i = t; i < H_ * D_ * F_; i += nth) {
    int h = i / (D_ * F_); int r = i % (D_ * F_);
    int d = r / F_;        int f = r % F_;
    WT[i] = (_Float16)W[(h * F_ + f) * D_ + d];
  }
  // WoT[c][k] = Wo[k][c]
  for (int i = t; i < C_ * HD_; i += nth) {
    int c = i / HD_; int k = i % HD_;
    WoT[i] = (_Float16)Wo[k * C_ + c];
  }
}

// ============================================================================
// K1: h = x @ W + Wb  (per b,h) -> hT f16 [B][H][D][N], sl/sr f32 [B][H][N]
// block = 8 waves, each wave: 16 rows x 64 cols; block covers 128 rows
// ============================================================================
__global__ __launch_bounds__(256) void k1_proj(const float* __restrict__ x,
                                               const _Float16* __restrict__ WT,
                                               const float* __restrict__ Wb,
                                               const float* __restrict__ a,
                                               _Float16* __restrict__ hT,
                                               float* __restrict__ sl,
                                               float* __restrict__ sr) {
  int blk  = blockIdx.x;            // ((b*H + h) * 8) + tile
  int tile = blk & 7;
  int bh   = blk >> 3;
  int h    = bh & 7;
  int b    = bh >> 3;
  int tid  = threadIdx.x;
  int w    = tid >> 5;
  int lane = tid & 31;
  int mrow = lane & 15, hi = lane >> 4;
  int n0   = tile * 128 + w * 16;

  v8f acc[4] = {};
  const float* xrow = x + ((size_t)(b * N_ + n0 + mrow)) * F_;
  for (int kk = 0; kk < F_; kk += 32) {
    v16h av;
#pragma unroll
    for (int e = 0; e < 16; ++e) {
      int k = hi * 16 + e;
      av[e] = (_Float16)xrow[kk + k];
    }
#pragma unroll
    for (int dt = 0; dt < 4; ++dt) {
      const _Float16* bp =
          WT + ((size_t)(h * D_ + dt * 16 + mrow)) * F_ + kk + hi * 16;
      v16h bv = *(const v16h*)bp;
      acc[dt] = wmma_f16(av, bv, acc[dt]);
    }
  }

  __shared__ float lt[8][16][68];   // per-wave 16x64 tile (+pad), f32, w/ bias
#pragma unroll
  for (int dt = 0; dt < 4; ++dt) {
    float bias = Wb[h * D_ + dt * 16 + mrow];
#pragma unroll
    for (int r = 0; r < 8; ++r)
      lt[w][r + hi * 8][dt * 16 + mrow] = acc[dt][r] + bias;
  }
  __syncthreads();

  // sl/sr: 2 threads per row (halves of D), pair-combine via lane shuffle
  {
    int row = tid >> 1, half = tid & 1;     // row in [0,128)
    int ww = row >> 4, m = row & 15;
    float pl = 0.f, pr = 0.f;
    for (int d = half * 32; d < half * 32 + 32; ++d) {
      float hv = lt[ww][m][d];
      pl += hv * a[h * 2 * D_ + d];
      pr += hv * a[h * 2 * D_ + D_ + d];
    }
    pl += __shfl_xor(pl, 1);
    pr += __shfl_xor(pr, 1);
    if (half == 0) {
      int n = tile * 128 + row;
      sl[(b * H_ + h) * N_ + n] = pl;
      sr[(b * H_ + h) * N_ + n] = pr;
    }
  }
  // hT[b][h][d][n] (f16, K-major for aggregation B-operand)
  {
    int d = tid >> 2, q = tid & 3;
    size_t base = ((size_t)((b * H_ + h) * D_ + d)) * N_ + tile * 128;
    for (int nn = 0; nn < 32; ++nn) {
      int nb = q * 32 + nn;
      hT[base + nb] = (_Float16)lt[nb >> 4][nb & 15][d];
    }
  }
}

// ============================================================================
// K2: fused masked-softmax attention + aggregation, layer 1.
// block = (b, 16-row i-tile); wave w = head w. attn matrix never materialized.
// ============================================================================
__global__ __launch_bounds__(256) void k2_attn1(const int* __restrict__ adj,
                                                const _Float16* __restrict__ hT,
                                                const float* __restrict__ sl,
                                                const float* __restrict__ sr,
                                                const float* __restrict__ ab,
                                                _Float16* __restrict__ zb) {
  int blk = blockIdx.x;             // b*64 + it
  int it  = blk & 63;
  int b   = blk >> 6;
  int i0  = it * 16;
  int tid = threadIdx.x;
  int w   = tid >> 5;
  int lane = tid & 31;
  int h = w;

  __shared__ unsigned mk[16][32];   // adjacency bitmask, 16 rows x 1024 bits
  __shared__ float srb[8][1024];    // sr rows, all heads
  __shared__ float stM[8][16];
  __shared__ float stIL[8][16];

  for (int wi = tid; wi < 512; wi += 256) {
    int row = wi >> 5, wd = wi & 31;
    const int* ap = adj + ((size_t)(b * N_ + i0 + row)) * N_ + wd * 32;
    unsigned bits = 0;
    for (int k = 0; k < 32; ++k) bits |= (ap[k] > 0 ? 1u : 0u) << k;
    mk[row][wd] = bits;
  }
  for (int i = tid; i < H_ * N_; i += 256)
    srb[i >> 10][i & 1023] = sr[(size_t)b * H_ * N_ + i];
  __syncthreads();

  float abh = ab[h];
  // ---- pass 1: per-row online softmax stats (max, sumexp) ----
  for (int m = 0; m < 16; ++m) {
    float slm = sl[(b * H_ + h) * N_ + i0 + m];
    float mrun = -1e30f, lrun = 0.f;
    for (int jc = 0; jc < 32; ++jc) {
      unsigned bits = mk[m][jc];
      if (bits & (1u << lane)) {
        float s = slm + srb[h][jc * 32 + lane] + abh;
        s = s >= 0.f ? s : ALPHA_ * s;
        float mn = fmaxf(mrun, s);
        lrun = lrun * __expf(mrun - mn) + __expf(s - mn);
        mrun = mn;
      }
    }
    for (int off = 16; off; off >>= 1) {
      float mo = __shfl_xor(mrun, off);
      float lo = __shfl_xor(lrun, off);
      float mn = fmaxf(mrun, mo);
      lrun = lrun * __expf(mrun - mn) + lo * __expf(mo - mn);
      mrun = mn;
    }
    if (lane == 0) { stM[w][m] = mrun; stIL[w][m] = 1.f / lrun; }
  }
  // same-wave LDS ops are in-order: safe to read back without barrier

  // ---- pass 2: P (on-the-fly, f16) @ hT via WMMA ----
  int mrow = lane & 15, hi = lane >> 4;
  float slr = sl[(b * H_ + h) * N_ + i0 + mrow];
  float mr = stM[w][mrow], il = stIL[w][mrow];
  v8f acc[4] = {};
  for (int j0 = 0; j0 < N_; j0 += 32) {
    v16h av;
#pragma unroll
    for (int e = 0; e < 16; ++e) {
      int k = hi * 16 + e;
      int j = j0 + k;
      unsigned bit = (mk[mrow][j >> 5] >> (j & 31)) & 1u;
      float s = slr + srb[h][j] + abh;
      s = s >= 0.f ? s : ALPHA_ * s;
      float p = bit ? __expf(s - mr) * il : 0.f;
      av[e] = (_Float16)p;
    }
#pragma unroll
    for (int dt = 0; dt < 4; ++dt) {
      const _Float16* bp =
          hT + ((size_t)((b * H_ + h) * D_ + dt * 16 + mrow)) * N_ + j0 + hi * 16;
      v16h bv = *(const v16h*)bp;
      acc[dt] = wmma_f16(av, bv, acc[dt]);
    }
  }
  // epilogue: ELU, store z (head-concat layout) as f16 A-operand for K3
#pragma unroll
  for (int dt = 0; dt < 4; ++dt) {
#pragma unroll
    for (int r = 0; r < 8; ++r) {
      float v = acc[dt][r];
      v = v > 0.f ? v : (__expf(v) - 1.f);
      int n = i0 + r + hi * 8;
      int d = dt * 16 + mrow;
      zb[((size_t)(b * N_ + n)) * HD_ + h * D_ + d] = (_Float16)v;
    }
  }
}

// ============================================================================
// K3: g = z @ Wo + Wob -> gT f16 [B][C][N]; tl/tr quarter-partials (no atomics)
// block = 32 rows; wave (w>>2)=row-half, (w&3)=64-col quarter
// ============================================================================
__global__ __launch_bounds__(256) void k3_outproj(const _Float16* __restrict__ zb,
                                                  const _Float16* __restrict__ WoT,
                                                  const float* __restrict__ Wob,
                                                  const float* __restrict__ ao,
                                                  _Float16* __restrict__ gT,
                                                  float* __restrict__ tlp,
                                                  float* __restrict__ trp) {
  int blk  = blockIdx.x;            // b*32 + tile
  int tile = blk & 31;
  int b    = blk >> 5;
  int tid  = threadIdx.x;
  int w    = tid >> 5;
  int lane = tid & 31;
  int mrow = lane & 15, hi = lane >> 4;
  int n0 = tile * 32 + (w >> 2) * 16;
  int cq = w & 3;
  int c0 = cq * 64;

  v8f acc[4] = {};
  const _Float16* zrow = zb + ((size_t)(b * N_ + n0 + mrow)) * HD_;
  for (int kk = 0; kk < HD_; kk += 32) {
    v16h av;
#pragma unroll
    for (int e = 0; e < 16; ++e) {
      int k = hi * 16 + e;
      av[e] = zrow[kk + k];
    }
#pragma unroll
    for (int ct = 0; ct < 4; ++ct) {
      const _Float16* bp =
          WoT + ((size_t)(c0 + ct * 16 + mrow)) * HD_ + kk + hi * 16;
      v16h bv = *(const v16h*)bp;
      acc[ct] = wmma_f16(av, bv, acc[ct]);
    }
  }
#pragma unroll
  for (int r = 0; r < 8; ++r) {
    int n = n0 + r + hi * 8;
    float pl = 0.f, pr = 0.f;
#pragma unroll
    for (int ct = 0; ct < 4; ++ct) {
      int c = c0 + ct * 16 + mrow;
      float g = acc[ct][r] + Wob[c];
      gT[((size_t)(b * C_ + c)) * N_ + n] = (_Float16)g;
      pl += g * ao[c];
      pr += g * ao[C_ + c];
    }
#pragma unroll
    for (int off = 1; off < 16; off <<= 1) {
      pl += __shfl_xor(pl, off);
      pr += __shfl_xor(pr, off);
    }
    if (mrow == 0) {   // lanes 0 and 16 each own 8 rows
      tlp[((size_t)(b * N_ + n)) * 4 + cq] = pl;
      trp[((size_t)(b * N_ + n)) * 4 + cq] = pr;
    }
  }
}

// ============================================================================
// K4: fused masked-softmax attention layer 2 + ELU + residual -> f32 out
// block = (b, 32-row tile); wave (w>>2)=row-half, (w&3)=64-col quarter
// ============================================================================
__global__ __launch_bounds__(256) void k4_attn2(const int* __restrict__ adj,
                                                const _Float16* __restrict__ gT,
                                                const float* __restrict__ tlp,
                                                const float* __restrict__ trp,
                                                const float* __restrict__ aob,
                                                const float* __restrict__ x,
                                                float* __restrict__ out) {
  int blk = blockIdx.x;             // b*32 + it
  int it  = blk & 31;
  int b   = blk >> 5;
  int i0  = it * 32;
  int tid = threadIdx.x;
  int w   = tid >> 5;
  int lane = tid & 31;

  __shared__ unsigned mk[32][32];
  __shared__ float trb[1024];
  __shared__ float tlb[32];
  __shared__ float stM[32];
  __shared__ float stIL[32];

  for (int wi = tid; wi < 1024; wi += 256) {
    int row = wi >> 5, wd = wi & 31;
    const int* ap = adj + ((size_t)(b * N_ + i0 + row)) * N_ + wd * 32;
    unsigned bits = 0;
    for (int k = 0; k < 32; ++k) bits |= (ap[k] > 0 ? 1u : 0u) << k;
    mk[row][wd] = bits;
  }
  for (int i = tid; i < 1024; i += 256) {
    const float* p = trp + ((size_t)(b * N_ + i)) * 4;
    trb[i] = p[0] + p[1] + p[2] + p[3];
  }
  if (tid < 32) {
    const float* p = tlp + ((size_t)(b * N_ + i0 + tid)) * 4;
    tlb[tid] = p[0] + p[1] + p[2] + p[3];
  }
  __syncthreads();

  float ab2 = aob[0];
  // ---- pass 1: stats; wave w owns rows 4w..4w+3 ----
  for (int mi = 0; mi < 4; ++mi) {
    int m = w * 4 + mi;
    float tlm = tlb[m];
    float mrun = -1e30f, lrun = 0.f;
    for (int jc = 0; jc < 32; ++jc) {
      unsigned bits = mk[m][jc];
      if (bits & (1u << lane)) {
        float s = tlm + trb[jc * 32 + lane] + ab2;
        s = s >= 0.f ? s : ALPHA_ * s;
        float mn = fmaxf(mrun, s);
        lrun = lrun * __expf(mrun - mn) + __expf(s - mn);
        mrun = mn;
      }
    }
    for (int off = 16; off; off >>= 1) {
      float mo = __shfl_xor(mrun, off);
      float lo = __shfl_xor(lrun, off);
      float mn = fmaxf(mrun, mo);
      lrun = lrun * __expf(mrun - mn) + lo * __expf(mo - mn);
      mrun = mn;
    }
    if (lane == 0) { stM[m] = mrun; stIL[m] = 1.f / lrun; }
  }
  __syncthreads();

  // ---- pass 2: P2 @ g via WMMA, 16 rows x 64 cols per wave ----
  int rg = w >> 2, c0 = (w & 3) * 64;
  int mrow = lane & 15, hi = lane >> 4;
  int irow = rg * 16 + mrow;
  float tlr = tlb[irow];
  float mr = stM[irow], il = stIL[irow];
  v8f acc[4] = {};
  for (int j0 = 0; j0 < N_; j0 += 32) {
    v16h av;
#pragma unroll
    for (int e = 0; e < 16; ++e) {
      int k = hi * 16 + e;
      int j = j0 + k;
      unsigned bit = (mk[irow][j >> 5] >> (j & 31)) & 1u;
      float s = tlr + trb[j] + ab2;
      s = s >= 0.f ? s : ALPHA_ * s;
      float p = bit ? __expf(s - mr) * il : 0.f;
      av[e] = (_Float16)p;
    }
#pragma unroll
    for (int ct = 0; ct < 4; ++ct) {
      const _Float16* bp =
          gT + ((size_t)(b * C_ + c0 + ct * 16 + mrow)) * N_ + j0 + hi * 16;
      v16h bv = *(const v16h*)bp;
      acc[ct] = wmma_f16(av, bv, acc[ct]);
    }
  }
  // epilogue: ELU + residual (C == F), f32 out
#pragma unroll
  for (int ct = 0; ct < 4; ++ct) {
#pragma unroll
    for (int r = 0; r < 8; ++r) {
      int n = i0 + rg * 16 + r + hi * 8;
      int c = c0 + ct * 16 + mrow;
      float v = acc[ct][r];
      v = v > 0.f ? v : (__expf(v) - 1.f);
      out[((size_t)(b * N_ + n)) * C_ + c] = v + x[((size_t)(b * N_ + n)) * F_ + c];
    }
  }
}

// ============================================================================
// Launch
// ============================================================================
extern "C" void kernel_launch(void* const* d_in, const int* in_sizes, int n_in,
                              void* d_out, int out_size, void* d_ws, size_t ws_size,
                              hipStream_t stream) {
  (void)in_sizes; (void)n_in; (void)out_size; (void)ws_size;
  const float* x   = (const float*)d_in[0];
  const int*   adj = (const int*)d_in[1];
  const float* W   = (const float*)d_in[2];
  const float* Wb  = (const float*)d_in[3];
  const float* a   = (const float*)d_in[4];
  const float* ab  = (const float*)d_in[5];
  const float* Wo  = (const float*)d_in[6];
  const float* Wob = (const float*)d_in[7];
  const float* ao  = (const float*)d_in[8];
  const float* aob = (const float*)d_in[9];
  float* out = (float*)d_out;

  char* ws = (char*)d_ws;
  _Float16* WT  = (_Float16*)(ws + 0);          //  H*D*F f16      = 256 KB
  _Float16* WoT = (_Float16*)(ws + 262144);     //  C*HD f16       = 256 KB
  _Float16* hT  = (_Float16*)(ws + 524288);     //  B*H*D*N f16    =   8 MB
  _Float16* zb  = (_Float16*)(ws + 8912896);    //  B*N*HD f16     =   8 MB
  _Float16* gT  = (_Float16*)(ws + 17301504);   //  B*C*N f16      =   4 MB
  float*    tlp = (float*)(ws + 21495808);      //  B*N*4 f32      = 128 KB
  float*    trp = (float*)(ws + 21626880);      //  B*N*4 f32      = 128 KB
  float*    sl  = (float*)(ws + 21757952);      //  B*H*N f32      = 256 KB
  float*    sr  = (float*)(ws + 22020096);      //  B*H*N f32      = 256 KB

  k0_prep   <<<256, 256, 0, stream>>>(W, Wo, WT, WoT);
  k1_proj   <<<512, 256, 0, stream>>>(x, WT, Wb, a, hT, sl, sr);
  k2_attn1  <<<512, 256, 0, stream>>>(adj, hT, sl, sr, ab, zb);
  k3_outproj<<<256, 256, 0, stream>>>(zb, WoT, Wob, ao, gT, tlp, trp);
  k4_attn2  <<<256, 256, 0, stream>>>(adj, gT, tlp, trp, aob, x, out);
}
